// DHSNN_81166291959855
// MI455X (gfx1250) — matrix operand order repaired and checked by reading, not compiled, not run on hardware
//
#include <hip/hip_runtime.h>

// ---------------- problem constants ----------------
#define T_STEPS 250
#define B_BATCH 128
#define F_IN    700
#define F_PAD   704            // K padded to multiple of 32
#define N_NEUR  512
#define NBR     2048           // NEURON_NUM * BRANCH
#define THRESH  0.3f
#define KCHUNKS (F_PAD / 32)   // 22
#define LDS_STRIDE 712         // F_PAD + 8: pads B rows to kill bank conflicts

typedef __attribute__((ext_vector_type(16))) __bf16        v16bf;
typedef __attribute__((ext_vector_type(8)))  float          v8f;
typedef __attribute__((ext_vector_type(16))) unsigned short v16us;

union Frag {
  v16us u;
  v16bf b;
  uint4 q[2];
};

__device__ __forceinline__ unsigned short f32_bf16_rne(float f) {
  unsigned int x = __float_as_uint(f);
  x += 0x7fffu + ((x >> 16) & 1u);
  return (unsigned short)(x >> 16);
}
__device__ __forceinline__ float bf16_hi_f32(unsigned short h) {
  return __uint_as_float(((unsigned int)h) << 16);
}
__device__ __forceinline__ float sigmoidf(float x) {
  return 1.0f / (1.0f + expf(-x));
}
// DPP quad_perm cross-lane add helper (branches live in one lane quad)
template <int CTRL>
__device__ __forceinline__ float quad_perm_f(float v) {
  int p = __builtin_amdgcn_update_dpp(0, __builtin_bit_cast(int, v),
                                      CTRL, 0xF, 0xF, true);
  return __builtin_bit_cast(float, p);
}

// ---------------------------------------------------------------------------
// Prep: split W (fp32 [2048 x 700]) -> bf16 hi/lo, K-padded to 704 with 0.
// ---------------------------------------------------------------------------
__global__ void prep_w_split(const float* __restrict__ W,
                             unsigned short* __restrict__ Whi,
                             unsigned short* __restrict__ Wlo) {
  int idx = blockIdx.x * blockDim.x + threadIdx.x;
  if (idx >= NBR * F_PAD) return;
  int row = idx / F_PAD;
  int k   = idx % F_PAD;
  float w = (k < F_IN) ? W[(size_t)row * F_IN + k] : 0.0f;
  unsigned short h = f32_bf16_rne(w);
  Whi[idx] = h;
  Wlo[idx] = f32_bf16_rne(w - bf16_hi_f32(h));
}

// ---------------------------------------------------------------------------
// Prep: split x (fp32 [T,B,700]) -> bf16 hi/lo planes padded to [T,B,704].
// Converted ONCE instead of 128x(col-groups) x 250x(steps) in the hot loop.
// ---------------------------------------------------------------------------
__global__ void prep_x_split(const float* __restrict__ x,
                             unsigned short* __restrict__ Xhi,
                             unsigned short* __restrict__ Xlo) {
  size_t idx = (size_t)blockIdx.x * blockDim.x + threadIdx.x;
  if (idx >= (size_t)T_STEPS * B_BATCH * F_PAD) return;
  size_t row = idx / F_PAD;
  int    k   = (int)(idx % F_PAD);
  float  f   = (k < F_IN) ? x[row * F_IN + k] : 0.0f;
  unsigned short h = f32_bf16_rne(f);
  Xhi[idx] = h;
  Xlo[idx] = f32_bf16_rne(f - bf16_hi_f32(h));
}

// ---------------------------------------------------------------------------
// Fused DH-SNN. Block = 64 batch rows x 16 pre-columns, 4 waves, each wave a
// 16x16 WMMA tile, looping all T with d/v/y state in registers.
// * GEMM: bf16x3 split (hi*hi + hi*lo + lo*hi) on v_wmma_f32_16x16x32_bf16.
// * B-panel (shared by all 4 waves, constant over t) cached in LDS once.
// * Branch-sum via DPP quad_perm (no LDS round-trip).
// ---------------------------------------------------------------------------
template <bool XPRE>
__global__ __launch_bounds__(128, 1) void dhsnn_fused(
    const float* __restrict__ tx,            // [T, B, F_IN] fp32
    const unsigned short* __restrict__ Xhi,  // [T, B, F_PAD] bf16 bits
    const unsigned short* __restrict__ Xlo,  // [T, B, F_PAD] bf16 bits
    const unsigned short* __restrict__ Whi,  // [NBR, F_PAD] bf16 bits
    const unsigned short* __restrict__ Wlo,  // [NBR, F_PAD] bf16 bits
    const float* __restrict__ bias,          // [NBR]
    const float* __restrict__ tau_m,         // [N_NEUR]
    const float* __restrict__ tau_n,         // [N_NEUR*4] flat == column index
    float* __restrict__ out)                 // [T, B, N_NEUR]
{
  __shared__ unsigned short sBhi[16 * LDS_STRIDE];   // 22.8 KB
  __shared__ unsigned short sBlo[16 * LDS_STRIDE];   // 22.8 KB

  const int lane  = threadIdx.x & 31;
  const int wave  = threadIdx.x >> 5;   // 0..3 row-groups inside block
  const int l16   = lane & 15;
  const int lhalf = lane >> 4;          // 0/1

  const int colbase = blockIdx.x * 16;              // output column tile
  const int rowbase = blockIdx.y * 64 + wave * 16;  // batch row tile

  const int col  = colbase + l16;   // pre column = n*4 + br (== tau_n flat idx)
  const int rowA = rowbase + l16;   // A-matrix row M for this lane

  // ---- cooperative fill of the block's B-panel into LDS (once) ----
  {
    const int tid = threadIdx.x;
    for (int i = tid; i < 16 * (F_PAD / 8); i += 128) {
      int g  = i % (F_PAD / 8);   // 8-ushort group within row
      int cc = i / (F_PAD / 8);   // column 0..15
      uint4 vh = *(const uint4*)(Whi + (size_t)(colbase + cc) * F_PAD + g * 8);
      uint4 vl = *(const uint4*)(Wlo + (size_t)(colbase + cc) * F_PAD + g * 8);
      *(uint4*)(&sBhi[cc * LDS_STRIDE + g * 8]) = vh;
      *(uint4*)(&sBlo[cc * LDS_STRIDE + g * 8]) = vl;
    }
  }

  // per-lane constants
  const float betav  = sigmoidf(tau_n[col]);
  const float omb    = 1.0f - betav;
  const float alphav = sigmoidf(tau_m[col >> 2]);
  const float bcol   = bias[col];

  __syncthreads();

  // B-fragment LDS base: lane holds K runs [8h..8h+7],[16+8h..23+8h] of W row `col`
  const unsigned short* sph = &sBhi[l16 * LDS_STRIDE + lhalf * 8];
  const unsigned short* spl = &sBlo[l16 * LDS_STRIDE + lhalf * 8];

  // persistent recurrence state (C-layout: VGPR r -> batch row rowC0 + r)
  float dstate[8], vstate[8], ystate[8];
#pragma unroll
  for (int r = 0; r < 8; ++r) { dstate[r] = 0.f; vstate[r] = 0.f; ystate[r] = 0.f; }

  const int  rowC0  = rowbase + lhalf * 8;
  const int  nOut   = col >> 2;
  const bool storer = ((lane & 3) == 0);   // one lane per branch quad stores y

  for (int t = 0; t < T_STEPS; ++t) {
    v8f acc = {0.f, 0.f, 0.f, 0.f, 0.f, 0.f, 0.f, 0.f};

    if (XPRE) {
      // A lane layout (16-bit A 16x32): lanes 0-15 hold K runs [0..7],[16..23];
      // lanes 16-31 hold [8..15],[24..31] -> base offset lhalf*8.
      const unsigned short* xh =
          Xhi + ((size_t)t * B_BATCH + rowA) * F_PAD + lhalf * 8;
      const unsigned short* xl =
          Xlo + ((size_t)t * B_BATCH + rowA) * F_PAD + lhalf * 8;

#pragma unroll 2
      for (int c = 0; c < KCHUNKS; ++c) {
        const int kbase = c * 32;

        Frag ahi, alo, bhi, blo;
        ahi.q[0] = *(const uint4*)(xh + kbase);
        ahi.q[1] = *(const uint4*)(xh + kbase + 16);
        alo.q[0] = *(const uint4*)(xl + kbase);
        alo.q[1] = *(const uint4*)(xl + kbase + 16);

        bhi.q[0] = *(const uint4*)(sph + kbase);
        bhi.q[1] = *(const uint4*)(sph + kbase + 16);
        blo.q[0] = *(const uint4*)(spl + kbase);
        blo.q[1] = *(const uint4*)(spl + kbase + 16);

        acc = __builtin_amdgcn_wmma_f32_16x16x32_bf16(false, ahi.b, false, bhi.b,
                                                      (short)0, acc, false, false);
        acc = __builtin_amdgcn_wmma_f32_16x16x32_bf16(false, ahi.b, false, blo.b,
                                                      (short)0, acc, false, false);
        acc = __builtin_amdgcn_wmma_f32_16x16x32_bf16(false, alo.b, false, bhi.b,
                                                      (short)0, acc, false, false);
      }
    } else {
      // fallback: convert fp32 x -> bf16 hi/lo on the fly
      const float* xrow = tx + ((size_t)t * B_BATCH + rowA) * F_IN;
#pragma unroll 1
      for (int c = 0; c < KCHUNKS; ++c) {
        const int kbase = c * 32;
        Frag ahi, alo;
        const int ka = kbase + lhalf * 8;
        if (c < KCHUNKS - 1) {
          float4 a0 = *(const float4*)(xrow + ka);
          float4 a1 = *(const float4*)(xrow + ka + 4);
          float4 a2 = *(const float4*)(xrow + ka + 16);
          float4 a3 = *(const float4*)(xrow + ka + 20);
          float fa[16] = {a0.x, a0.y, a0.z, a0.w, a1.x, a1.y, a1.z, a1.w,
                          a2.x, a2.y, a2.z, a2.w, a3.x, a3.y, a3.z, a3.w};
#pragma unroll
          for (int e = 0; e < 16; ++e) {
            unsigned short h = f32_bf16_rne(fa[e]);
            ahi.u[e] = h;
            alo.u[e] = f32_bf16_rne(fa[e] - bf16_hi_f32(h));
          }
        } else {
#pragma unroll
          for (int e = 0; e < 16; ++e) {
            int   k = ka + ((e < 8) ? e : e + 8);
            float f = (k < F_IN) ? xrow[k] : 0.0f;
            unsigned short h = f32_bf16_rne(f);
            ahi.u[e] = h;
            alo.u[e] = f32_bf16_rne(f - bf16_hi_f32(h));
          }
        }
        Frag bhi, blo;
        bhi.q[0] = *(const uint4*)(sph + kbase);
        bhi.q[1] = *(const uint4*)(sph + kbase + 16);
        blo.q[0] = *(const uint4*)(spl + kbase);
        blo.q[1] = *(const uint4*)(spl + kbase + 16);

        acc = __builtin_amdgcn_wmma_f32_16x16x32_bf16(false, ahi.b, false, bhi.b,
                                                      (short)0, acc, false, false);
        acc = __builtin_amdgcn_wmma_f32_16x16x32_bf16(false, ahi.b, false, blo.b,
                                                      (short)0, acc, false, false);
        acc = __builtin_amdgcn_wmma_f32_16x16x32_bf16(false, alo.b, false, bhi.b,
                                                      (short)0, acc, false, false);
      }
    }

    // ---- recurrence + spike (elementwise on the C tile) ----
    float yarr[8];
#pragma unroll
    for (int r = 0; r < 8; ++r) {
      float pre = acc[r] + bcol;
      float dn  = betav * dstate[r] + omb * pre;
      dstate[r] = dn;
      // branch sum across the lane quad via DPP quad_perm (xor1 then xor2)
      float s = dn + quad_perm_f<0xB1>(dn);   // [1,0,3,2]
      s += quad_perm_f<0x4E>(s);              // [2,3,0,1]
      float vv = vstate[r] * alphav + s - THRESH * ystate[r];
      float yy = (vv > THRESH) ? 1.0f : 0.0f;
      vstate[r] = vv;
      ystate[r] = yy;
      yarr[r] = yy;
    }
    if (storer) {
      float* p = out + ((size_t)t * B_BATCH + rowC0) * N_NEUR + nOut;
#pragma unroll
      for (int r = 0; r < 8; ++r)
        p[(size_t)r * N_NEUR] = yarr[r];   // immediate-offset stores
    }
  }
}

// ---------------------------------------------------------------------------
extern "C" void kernel_launch(void* const* d_in, const int* in_sizes, int n_in,
                              void* d_out, int out_size, void* d_ws, size_t ws_size,
                              hipStream_t stream) {
  const float* tx    = (const float*)d_in[0];
  const float* W     = (const float*)d_in[1];
  const float* bias  = (const float*)d_in[2];
  const float* tau_m = (const float*)d_in[3];
  const float* tau_n = (const float*)d_in[4];

  const size_t wElems = (size_t)NBR * F_PAD;                  // 1.44M
  const size_t xElems = (size_t)T_STEPS * B_BATCH * F_PAD;    // 22.5M
  const size_t needW  = 2 * wElems * sizeof(unsigned short);  // 5.8 MB
  const size_t needX  = 2 * xElems * sizeof(unsigned short);  // 90.1 MB

  unsigned short* Whi = (unsigned short*)d_ws;
  unsigned short* Wlo = Whi + wElems;
  unsigned short* Xhi = Wlo + wElems;
  unsigned short* Xlo = Xhi + xElems;

  prep_w_split<<<(int)((wElems + 255) / 256), 256, 0, stream>>>(W, Whi, Wlo);

  dim3 grid(NBR / 16, B_BATCH / 64);   // 128 x 2 = 256 blocks, 4 waves each

  if (ws_size >= needW + needX) {
    prep_x_split<<<(int)((xElems + 255) / 256), 256, 0, stream>>>(tx, Xhi, Xlo);
    dhsnn_fused<true><<<grid, 128, 0, stream>>>(tx, Xhi, Xlo, Whi, Wlo, bias,
                                                tau_m, tau_n, (float*)d_out);
  } else {
    dhsnn_fused<false><<<grid, 128, 0, stream>>>(tx, (const unsigned short*)nullptr,
                                                 (const unsigned short*)nullptr,
                                                 Whi, Wlo, bias, tau_m, tau_n,
                                                 (float*)d_out);
  }
}
